// dfm_criteo_70935679861554
// MI455X (gfx1250) — compile-verified, matrix operand order checked by reading, not run on hardware
//
#include <hip/hip_runtime.h>
#include <hip/hip_bf16.h>
#include <math.h>

// ---------------- types ----------------
typedef __attribute__((ext_vector_type(16))) __bf16 v16bf;
typedef __attribute__((ext_vector_type(8)))  __bf16 v8bf;
typedef __attribute__((ext_vector_type(8)))  float  v8f;

#define B_BATCH   16384
#define EMB       128
#define N_SPARSE  26
#define N_DENSE   13
#define HID       256
#define K1        (N_SPARSE * EMB)   // 3328

// f32 -> bf16 round-to-nearest-even
__device__ __forceinline__ __bf16 f2bf(float f) {
    union { float f; unsigned u; } v; v.f = f;
    unsigned r = v.u + 0x7FFFu + ((v.u >> 16) & 1u);
    unsigned short h = (unsigned short)(r >> 16);
    return __builtin_bit_cast(__bf16, h);
}
__device__ __forceinline__ float bf2f(__bf16 b) {
    unsigned short h = __builtin_bit_cast(unsigned short, b);
    unsigned u = ((unsigned)h) << 16;
    return __builtin_bit_cast(float, u);
}

__device__ __forceinline__ v16bf pack16(v8bf lo, v8bf hi) {
    return __builtin_shufflevector(lo, hi, 0,1,2,3,4,5,6,7,8,9,10,11,12,13,14,15);
}

__device__ __forceinline__ v8f wmma_bf16(v16bf a, v16bf b, v8f c) {
    // (neg_a, A, neg_b, B, c_mod, C, reuse_a, reuse_b)
    return __builtin_amdgcn_wmma_f32_16x16x32_bf16(false, a, false, b, (short)0, c, false, false);
}

// ---------------- kernel 1: weight conversion f32 -> bf16 ----------------
__global__ void dfm_cvt_kernel(const float* __restrict__ w1, const float* __restrict__ w2,
                               __bf16* __restrict__ w1bf, __bf16* __restrict__ w2bf) {
    int i = blockIdx.x * blockDim.x + threadIdx.x;
    const int N1 = HID * K1;          // 851968
    const int N2 = HID * HID;         // 65536
    if (i < N1) {
        w1bf[i] = f2bf(w1[i]);
    } else {
        int j = i - N1;
        if (j < N2) w2bf[j] = f2bf(w2[j]);
    }
}

// ---------------- kernel 2: FM first + second order, one wave per sample ----------------
__global__ void dfm_fm_kernel(const float* __restrict__ dense, const int* __restrict__ sparse,
                              const float* __restrict__ emb1, const float* __restrict__ emb2,
                              const float* __restrict__ fm_w, float* __restrict__ y12) {
    const int wv = threadIdx.x >> 5;
    const int ln = threadIdx.x & 31;
    const int b  = blockIdx.x * 8 + wv;

    float s0=0.f,s1=0.f,s2=0.f,s3=0.f, q0=0.f,q1=0.f,q2=0.f,q3=0.f;
    #pragma unroll
    for (int f = 0; f < N_SPARSE; ++f) {
        int idx = sparse[b * N_SPARSE + f];
        const float4 e = *(const float4*)(emb2 + (size_t)idx * EMB + ln * 4);
        s0 += e.x; q0 += e.x * e.x;
        s1 += e.y; q1 += e.y * e.y;
        s2 += e.z; q2 += e.z * e.z;
        s3 += e.w; q3 += e.w * e.w;
    }
    float p = 0.5f * ((s0*s0 - q0) + (s1*s1 - q1) + (s2*s2 - q2) + (s3*s3 - q3));
    if (ln < N_SPARSE) p += emb1[sparse[b * N_SPARSE + ln]];
    if (ln < N_DENSE)  p += dense[b * N_DENSE + ln] * fm_w[ln];
    #pragma unroll
    for (int off = 16; off >= 1; off >>= 1) p += __shfl_xor(p, off, 32);
    if (ln == 0) y12[b] = p;
}

// ---------------- kernel 3: GEMM1 (fused gather) h1 = relu(flat @ w1^T), bf16 out ----------
// block = 256 threads = 8 waves; block owns 32 rows x 256 cols of h1.
// Each wave: 2 M-tiles x 2 N-tiles of 16x16 -> every fragment feeds 2 WMMAs.
__global__ void dfm_gemm1_kernel(const int* __restrict__ sparse, const float* __restrict__ emb2,
                                 const __bf16* __restrict__ w1bf, __bf16* __restrict__ h1bf) {
    __shared__ __bf16 aT[32 * 136];        // 32 rows, K-chunk 128, stride 136 (bank de-pattern)
    __shared__ int    sIdx[32 * N_SPARSE]; // 832 indices

    const int m0 = blockIdx.x * 32;
    const int t  = threadIdx.x;
    const int wv = t >> 5;
    const int ln = t & 31;

    #pragma unroll
    for (int u = t; u < 32 * N_SPARSE; u += 256) sIdx[u] = sparse[m0 * N_SPARSE + u];
    __syncthreads();

    v8f acc00 = {}, acc01 = {}, acc10 = {}, acc11 = {};

    const int mA  = ln & 15;              // A-frag row within tile
    const int kbA = (ln < 16) ? 0 : 8;    // A/B-frag K sub-offset per documented layout
    const int nCol = wv * 32 + (ln & 15); // first N-tile column for this lane

    for (int f = 0; f < N_SPARSE; ++f) {
        // ---- stage 32x128 f32 gather -> bf16 in LDS (2 units of 8 dims per thread) ----
        #pragma unroll
        for (int u = 0; u < 2; ++u) {
            const int unit = t + u * 256;       // 0..511
            const int rs = unit >> 4;           // row 0..31
            const int sg = unit & 15;           // 8-dim segment
            int idx = sIdx[rs * N_SPARSE + f];
            const float4* src = (const float4*)(emb2 + (size_t)idx * EMB + sg * 8);
            float4 a0 = src[0], a1 = src[1];
            v8bf pk;
            pk[0]=f2bf(a0.x); pk[1]=f2bf(a0.y); pk[2]=f2bf(a0.z); pk[3]=f2bf(a0.w);
            pk[4]=f2bf(a1.x); pk[5]=f2bf(a1.y); pk[6]=f2bf(a1.z); pk[7]=f2bf(a1.w);
            *(v8bf*)(&aT[rs * 136 + sg * 8]) = pk;
        }
        __syncthreads();

        // ---- 4 K-steps of 32; 2x2 tiles per wave ----
        #pragma unroll
        for (int j = 0; j < 4; ++j) {
            const __bf16* ap0 = &aT[mA * 136 + j * 32 + kbA];
            v16bf a0 = pack16(*(const v8bf*)ap0, *(const v8bf*)(ap0 + 16));
            const __bf16* ap1 = ap0 + 16 * 136;
            v16bf a1 = pack16(*(const v8bf*)ap1, *(const v8bf*)(ap1 + 16));

            const int kg = f * EMB + j * 32 + kbA;
            const __bf16* bp0 = w1bf + (size_t)nCol * K1 + kg;
            v16bf b0 = pack16(*(const v8bf*)bp0, *(const v8bf*)(bp0 + 16));
            const __bf16* bp1 = bp0 + (size_t)16 * K1;
            v16bf b1 = pack16(*(const v8bf*)bp1, *(const v8bf*)(bp1 + 16));

            acc00 = wmma_bf16(a0, b0, acc00);
            acc01 = wmma_bf16(a0, b1, acc01);
            acc10 = wmma_bf16(a1, b0, acc10);
            acc11 = wmma_bf16(a1, b1, acc11);
        }
        __syncthreads();
    }

    // ---- epilogue: relu -> bf16, C layout: M = r + 8*(lane>=16), N = lane&15 ----
    const int mC = (ln < 16) ? 0 : 8;
    const int nC = ln & 15;
    #pragma unroll
    for (int r = 0; r < 8; ++r) {
        float v;
        int row0 = m0 + mC + r, row1 = row0 + 16;
        v = acc00[r]; v = v > 0.f ? v : 0.f; h1bf[(size_t)row0 * HID + wv * 32 + nC]      = f2bf(v);
        v = acc01[r]; v = v > 0.f ? v : 0.f; h1bf[(size_t)row0 * HID + wv * 32 + 16 + nC] = f2bf(v);
        v = acc10[r]; v = v > 0.f ? v : 0.f; h1bf[(size_t)row1 * HID + wv * 32 + nC]      = f2bf(v);
        v = acc11[r]; v = v > 0.f ? v : 0.f; h1bf[(size_t)row1 * HID + wv * 32 + 16 + nC] = f2bf(v);
    }
}

// ---------------- kernel 4: GEMM2 h2 = relu(h1 @ w2^T), bf16; 32 rows/block ------------
__global__ void dfm_gemm2_kernel(const __bf16* __restrict__ h1bf, const __bf16* __restrict__ w2bf,
                                 __bf16* __restrict__ h2bf) {
    const int m0 = blockIdx.x * 32;
    const int t  = threadIdx.x;
    const int wv = t >> 5;
    const int ln = t & 31;

    v8f acc00 = {}, acc01 = {}, acc10 = {}, acc11 = {};
    const int mA  = ln & 15;
    const int kbA = (ln < 16) ? 0 : 8;
    const int nCol = wv * 32 + (ln & 15);

    #pragma unroll
    for (int kt = 0; kt < HID / 32; ++kt) {
        const int kb = kt * 32 + kbA;
        const __bf16* ap0 = h1bf + (size_t)(m0 + mA) * HID + kb;
        v16bf a0 = pack16(*(const v8bf*)ap0, *(const v8bf*)(ap0 + 16));
        const __bf16* ap1 = ap0 + (size_t)16 * HID;
        v16bf a1 = pack16(*(const v8bf*)ap1, *(const v8bf*)(ap1 + 16));

        const __bf16* bp0 = w2bf + (size_t)nCol * HID + kb;
        v16bf b0 = pack16(*(const v8bf*)bp0, *(const v8bf*)(bp0 + 16));
        const __bf16* bp1 = bp0 + (size_t)16 * HID;
        v16bf b1 = pack16(*(const v8bf*)bp1, *(const v8bf*)(bp1 + 16));

        acc00 = wmma_bf16(a0, b0, acc00);
        acc01 = wmma_bf16(a0, b1, acc01);
        acc10 = wmma_bf16(a1, b0, acc10);
        acc11 = wmma_bf16(a1, b1, acc11);
    }

    const int mC = (ln < 16) ? 0 : 8;
    const int nC = ln & 15;
    #pragma unroll
    for (int r = 0; r < 8; ++r) {
        float v;
        int row0 = m0 + mC + r, row1 = row0 + 16;
        v = acc00[r]; v = v > 0.f ? v : 0.f; h2bf[(size_t)row0 * HID + wv * 32 + nC]      = f2bf(v);
        v = acc01[r]; v = v > 0.f ? v : 0.f; h2bf[(size_t)row0 * HID + wv * 32 + 16 + nC] = f2bf(v);
        v = acc10[r]; v = v > 0.f ? v : 0.f; h2bf[(size_t)row1 * HID + wv * 32 + nC]      = f2bf(v);
        v = acc11[r]; v = v > 0.f ? v : 0.f; h2bf[(size_t)row1 * HID + wv * 32 + 16 + nC] = f2bf(v);
    }
}

// ---------------- kernel 5: y = sigmoid(y12 + h2 . w3), one wave per sample ------------
__global__ void dfm_final_kernel(const __bf16* __restrict__ h2bf, const float* __restrict__ w3,
                                 const float* __restrict__ y12, float* __restrict__ out) {
    const int wv = threadIdx.x >> 5;
    const int ln = threadIdx.x & 31;
    const int b  = blockIdx.x * 8 + wv;

    const v8bf h = *(const v8bf*)(h2bf + (size_t)b * HID + ln * 8);
    float dot = 0.f;
    #pragma unroll
    for (int i = 0; i < 8; ++i) dot += bf2f(h[i]) * w3[ln * 8 + i];
    #pragma unroll
    for (int off = 16; off >= 1; off >>= 1) dot += __shfl_xor(dot, off, 32);
    if (ln == 0) {
        float x = y12[b] + dot;
        out[b] = 1.0f / (1.0f + __expf(-x));
    }
}

// ---------------- host launcher ----------------
extern "C" void kernel_launch(void* const* d_in, const int* in_sizes, int n_in,
                              void* d_out, int out_size, void* d_ws, size_t ws_size,
                              hipStream_t stream) {
    const float* dense  = (const float*)d_in[0];
    const int*   sparse = (const int*)  d_in[1];
    const float* emb1   = (const float*)d_in[2];
    const float* emb2   = (const float*)d_in[3];
    const float* fm_w   = (const float*)d_in[4];
    const float* w1     = (const float*)d_in[5];
    const float* w2     = (const float*)d_in[6];
    const float* w3     = (const float*)d_in[7];
    float* out = (float*)d_out;

    char* ws = (char*)d_ws;
    // workspace layout (bytes), all 16B aligned
    __bf16* w1bf = (__bf16*)(ws + 0);                    // 256*3328*2 = 1,703,936
    __bf16* w2bf = (__bf16*)(ws + 1703936);              // 256*256*2  =   131,072
    __bf16* h1bf = (__bf16*)(ws + 1835008);              // 16384*256*2 = 8,388,608
    __bf16* h2bf = (__bf16*)(ws + 10223616);             // 8,388,608
    float*  y12  = (float*)(ws + 18612224);              // 16384*4

    // 1) weights -> bf16 (917504 elems)
    dfm_cvt_kernel<<<3584, 256, 0, stream>>>(w1, w2, w1bf, w2bf);
    // 2) FM part -> y12
    dfm_fm_kernel<<<B_BATCH / 8, 256, 0, stream>>>(dense, sparse, emb1, emb2, fm_w, y12);
    // 3) GEMM1 with fused embedding gather (32 rows per block)
    dfm_gemm1_kernel<<<B_BATCH / 32, 256, 0, stream>>>(sparse, emb2, w1bf, h1bf);
    // 4) GEMM2 (32 rows per block)
    dfm_gemm2_kernel<<<B_BATCH / 32, 256, 0, stream>>>(h1bf, w2bf, h2bf);
    // 5) final dot + sigmoid
    dfm_final_kernel<<<B_BATCH / 8, 256, 0, stream>>>(h2bf, w3, y12, out);
}
